// EPIField_66838281060618
// MI455X (gfx1250) — matrix operand order repaired and checked by reading, not compiled, not run on hardware
//
#include <hip/hip_runtime.h>
#include <hip/hip_bf16.h>

typedef __attribute__((ext_vector_type(16))) _Float16 v16h;
typedef __attribute__((ext_vector_type(8)))  _Float16 v8h;
typedef __attribute__((ext_vector_type(8)))  float    v8f;

#define N_TOT 524288
#define MROWS 64

// Padded-K weight layout in workspace (f16, transposed: [256][Kpad] row-major)
constexpr size_t O0 = 0;                     // L0: K 168 -> 192
constexpr size_t O1 = O0 + 256 * 192;        // L1: 256
constexpr size_t O2 = O1 + 256 * 256;        // L2
constexpr size_t O3 = O2 + 256 * 256;        // L3
constexpr size_t O4 = O3 + 256 * 256;        // L4: K 424 -> 448
constexpr size_t O5 = O4 + 256 * 448;        // L5
constexpr size_t O6 = O5 + 256 * 256;        // L6
constexpr size_t O7 = O6 + 256 * 256;        // L7
constexpr size_t OC = O7 + 256 * 256;        // Wc: K 268 -> 288

// ---------------- weight transpose/convert pre-pass ----------------
__global__ void conv_w_kernel(const float* __restrict__ W,
                              _Float16* __restrict__ dst,
                              int kreal, int kpad) {
    int idx = blockIdx.x * blockDim.x + threadIdx.x;
    int total = 256 * kpad;
    if (idx >= total) return;
    int n = idx / kpad;
    int k = idx - n * kpad;
    float v = (k < kreal) ? W[(size_t)k * 256 + n] : 0.0f;
    dst[idx] = (_Float16)v;
}

// ---------------- main fused MLP kernel ----------------
struct EpiParams {
    const float* uv;
    const float* st;
    const float* epi;
    const float* dist;
    const _Float16* wt;     // workspace f16 weights
    const float* b[9];      // b0..b7, bc
    const float* Wa;
    const float* ba;
    const float* Wo;
    const float* bo;
    float* out;
};

// A operand: lane m = lane&15 (row), h = lane>>4.
// halves 0..7  = A[m][kb + h*8 + 0..7], halves 8..15 = A[m][kb + 16 + h*8 + 0..7]
__device__ __forceinline__ v16h load_a(const _Float16* p) {
    union { v16h v; v8h h[2]; } u;
    u.h[0] = *(const v8h*)(p);
    u.h[1] = *(const v8h*)(p + 16);
    return u.v;
}

// One layer: dst[64][256] = relu( [src0 | src1] @ Wt^T + bias ), all WMMA f16->f32.
// Wave w: rows (w&3)*16..+15, cols (w>>2)*128..+127, processed as two 64-col
// passes of 4 accumulator tiles each. Explicit 1-deep software pipeline: the
// next k-step's A (LDS) and 4 B tiles (global) are fetched into fresh values
// before the current step's 4 WMMAs issue, so loads stay in flight across
// each WMMA group (k-loops fully unroll: ks0/ks1 are literal at call sites).
__device__ __forceinline__ void mlp_layer(
    const _Float16* __restrict__ s0, int str0, int ks0,
    const _Float16* __restrict__ s1, int str1, int ks1,
    const _Float16* __restrict__ wt, int kpad,
    const float* __restrict__ bias,
    _Float16* __restrict__ dst) {
    const int tid  = threadIdx.x;
    const int lane = tid & 31;
    const int wv   = tid >> 5;
    const int r0   = (wv & 3) * 16;    // row-tile of this wave
    const int c0   = (wv >> 2) * 128;  // 128-column slice of this wave
    const int m    = lane & 15;
    const int h    = lane >> 4;
    const int kst  = ks0 + ks1;

    for (int nc = 0; nc < 2; ++nc) {
        const int cbase = c0 + nc * 64;
        v8f acc[4] = {v8f{0.f}, v8f{0.f}, v8f{0.f}, v8f{0.f}};
        // B operand base: lane n = m, K = h*8+0..7 and 16+h*8+0..7 (matches A)
        const _Float16* wrow = wt + (size_t)(cbase + m) * kpad + h * 8;

        // k-step -> A pointer (two concatenated LDS sources)
        auto aptr = [&](int ks) -> const _Float16* {
            return (ks < ks0)
                ? (s0 + (r0 + m) * str0 + ks * 32 + h * 8)
                : (s1 + (r0 + m) * str1 + (ks - ks0) * 32 + h * 8);
        };

        // prologue: fetch k-step 0
        v16h acur = load_a(aptr(0));
        v16h bcur[4], bnxt[4];
        v16h anxt;
#pragma unroll
        for (int t = 0; t < 4; ++t)
            bcur[t] = load_a(wrow + (size_t)t * 16 * kpad);

        for (int ks = 0; ks < kst; ++ks) {
            if (ks + 1 < kst) {
                anxt = load_a(aptr(ks + 1));
                const _Float16* wp = wrow + (ks + 1) * 32;
#pragma unroll
                for (int t = 0; t < 4; ++t)
                    bnxt[t] = load_a(wp + (size_t)t * 16 * kpad);
            }
#pragma unroll
            for (int t = 0; t < 4; ++t)
                acc[t] = __builtin_amdgcn_wmma_f32_16x16x32_f16(
                    false, acur, false, bcur[t], (short)0, acc[t], false, false);
            acur = anxt;
#pragma unroll
            for (int t = 0; t < 4; ++t)
                bcur[t] = bnxt[t];
        }

#pragma unroll
        for (int t = 0; t < 4; ++t) {
            const int col = cbase + t * 16 + m;
            const float bv = bias[col];
#pragma unroll
            for (int v = 0; v < 8; ++v) {
                float x = acc[t][v] + bv;       // C/D: row r0+h*8+v, col = col
                x = x > 0.f ? x : 0.f;
                dst[(r0 + h * 8 + v) * 256 + col] = (_Float16)x;
            }
        }
    }
}

__global__ __launch_bounds__(256)
void epifield_kernel(EpiParams p) {
    __shared__ __attribute__((aligned(32))) _Float16 emb[MROWS * 192];
    __shared__ __attribute__((aligned(32))) _Float16 uvt[MROWS * 32];
    __shared__ __attribute__((aligned(32))) _Float16 xa[MROWS * 256];
    __shared__ __attribute__((aligned(32))) _Float16 xb[MROWS * 256];

    const int tid = threadIdx.x;
    const int R   = blockIdx.x * MROWS;

    // Stage inputs: all_embeds = [st | epi] zero-padded 168->192, uv padded 12->32
    for (int i = tid; i < MROWS * 192; i += 256) {
        int r = i / 192, c = i - r * 192;
        float v = 0.f;
        if (c < 84)        v = p.st[(size_t)(R + r) * 84 + c];
        else if (c < 168)  v = p.epi[(size_t)(R + r) * 84 + (c - 84)];
        emb[i] = (_Float16)v;
    }
    for (int i = tid; i < MROWS * 32; i += 256) {
        int r = i / 32, c = i - r * 32;
        float v = (c < 12) ? p.uv[(size_t)(R + r) * 12 + c] : 0.f;
        uvt[i] = (_Float16)v;
    }
    __syncthreads();

    mlp_layer(emb, 192, 6, nullptr, 0, 0, p.wt + O0, 192, p.b[0], xa); __syncthreads();
    mlp_layer(xa, 256, 8, nullptr, 0, 0, p.wt + O1, 256, p.b[1], xb); __syncthreads();
    mlp_layer(xb, 256, 8, nullptr, 0, 0, p.wt + O2, 256, p.b[2], xa); __syncthreads();
    mlp_layer(xa, 256, 8, nullptr, 0, 0, p.wt + O3, 256, p.b[3], xb); __syncthreads();
    // skip concat: [x(256) | all_embeds(192pad)]
    mlp_layer(xb, 256, 8, emb, 192, 6, p.wt + O4, 448, p.b[4], xa); __syncthreads();
    mlp_layer(xa, 256, 8, nullptr, 0, 0, p.wt + O5, 256, p.b[5], xb); __syncthreads();
    mlp_layer(xb, 256, 8, nullptr, 0, 0, p.wt + O6, 256, p.b[6], xa); __syncthreads();
    mlp_layer(xa, 256, 8, nullptr, 0, 0, p.wt + O7, 256, p.b[7], xb); __syncthreads();
    // color trunk: [x(256) | uv(32pad)] -> h in xa ; final x stays in xb
    mlp_layer(xb, 256, 8, uvt, 32, 1, p.wt + OC, 288, p.b[8], xa); __syncthreads();

    // alpha head: pred_alpha = x @ Wa + ba ; alpha = 1 - exp(-relu(pa)*dist)
    if (tid < MROWS) {
        const int r = tid;
        const size_t row = (size_t)R + r;
        float pa = p.ba[0];
        const _Float16* xr = xb + r * 256;
        for (int k = 0; k < 256; ++k) pa += (float)xr[k] * p.Wa[k];
        pa = pa > 0.f ? pa : 0.f;
        p.out[row] = 1.0f - __expf(-pa * p.dist[row]);
    }
    // color head: sigmoid(h @ Wo + bo) * 1.002 - 0.001
    if (tid < MROWS * 3) {
        const int r = tid / 3, j = tid - (tid / 3) * 3;
        const size_t row = (size_t)R + r;
        float z = p.bo[j];
        const _Float16* hr = xa + r * 256;
        for (int k = 0; k < 256; ++k) z += (float)hr[k] * p.Wo[k * 3 + j];
        float s = 1.0f / (1.0f + __expf(-z));
        p.out[(size_t)N_TOT + row * 3 + j] = s * 1.002f - 0.001f;
    }
}

// ---------------- launch ----------------
extern "C" void kernel_launch(void* const* d_in, const int* in_sizes, int n_in,
                              void* d_out, int out_size, void* d_ws, size_t ws_size,
                              hipStream_t stream) {
    (void)in_sizes; (void)n_in; (void)out_size; (void)ws_size;

    _Float16* wt = (_Float16*)d_ws;

    // d_in order: uv, st, epi, dist, then params dict: W0,b0,...,W7,b7,Wa,ba,Wc,bc,Wo,bo
    struct LayerW { const float* W; size_t off; int kreal, kpad; };
    const LayerW lw[9] = {
        {(const float*)d_in[4],  O0, 168, 192},
        {(const float*)d_in[6],  O1, 256, 256},
        {(const float*)d_in[8],  O2, 256, 256},
        {(const float*)d_in[10], O3, 256, 256},
        {(const float*)d_in[12], O4, 424, 448},
        {(const float*)d_in[14], O5, 256, 256},
        {(const float*)d_in[16], O6, 256, 256},
        {(const float*)d_in[18], O7, 256, 256},
        {(const float*)d_in[22], OC, 268, 288},
    };
    for (int i = 0; i < 9; ++i) {
        int total = 256 * lw[i].kpad;
        conv_w_kernel<<<(total + 255) / 256, 256, 0, stream>>>(
            lw[i].W, wt + lw[i].off, lw[i].kreal, lw[i].kpad);
    }

    EpiParams P;
    P.uv   = (const float*)d_in[0];
    P.st   = (const float*)d_in[1];
    P.epi  = (const float*)d_in[2];
    P.dist = (const float*)d_in[3];
    P.wt   = wt;
    P.b[0] = (const float*)d_in[5];
    P.b[1] = (const float*)d_in[7];
    P.b[2] = (const float*)d_in[9];
    P.b[3] = (const float*)d_in[11];
    P.b[4] = (const float*)d_in[13];
    P.b[5] = (const float*)d_in[15];
    P.b[6] = (const float*)d_in[17];
    P.b[7] = (const float*)d_in[19];
    P.b[8] = (const float*)d_in[23];   // bc
    P.Wa   = (const float*)d_in[20];
    P.ba   = (const float*)d_in[21];
    P.Wo   = (const float*)d_in[24];
    P.bo   = (const float*)d_in[25];
    P.out  = (float*)d_out;

    epifield_kernel<<<N_TOT / MROWS, 256, 0, stream>>>(P);
}
